// Net_35725537968348
// MI455X (gfx1250) — compile-verified, matrix operand order checked by reading, not run on hardware
//
#include <hip/hip_runtime.h>
#include <math.h>

// Problem constants (match reference)
#define NN 50000
#define EE 800000
#define BB 500
#define FF 128
#define HH 256
#define AA 16
#define KKR 16   // routing kernels
#define NHD 4
#define CC 10

// Padded row counts (multiples of BM) so GEMM staging needs no row guards.
#define NPAD 50048
#define BPAD 512

// GEMM tiling
#define BM 64
#define BN 128
#define BK 32

typedef __attribute__((ext_vector_type(16))) _Float16 v16h;
typedef __attribute__((ext_vector_type(8)))  _Float16 v8h;
typedef __attribute__((ext_vector_type(8)))  float    v8f;

// ---------------------------------------------------------------------------
// Optional CDNA5 async global->LDS copy (GLOBAL_LOAD_ASYNC_TO_LDS_B128,
// tracked by ASYNCcnt). Falls back to a synchronous b128 copy when the
// builtins are not exposed by the toolchain.
// The builtin's prototype (from the compiler diagnostic) takes a pointer to
// a GCC-vector int4 in the global (AS1) address space, then the LDS (AS3)
// destination, then two immediates (offset, cpol).
// ---------------------------------------------------------------------------
#if __has_builtin(__builtin_amdgcn_global_load_async_to_lds_b128) && \
    __has_builtin(__builtin_amdgcn_s_wait_asynccnt)
#define HAS_ASYNC_LDS 1
#endif

typedef int v4i_gcc __attribute__((vector_size(16)));

__device__ __forceinline__ void cp16_g2l(_Float16* lds, const _Float16* g) {
#ifdef HAS_ASYNC_LDS
  __builtin_amdgcn_global_load_async_to_lds_b128(
      (__attribute__((address_space(1))) v4i_gcc*)g,
      (__attribute__((address_space(3))) v4i_gcc*)lds, 0, 0);
#else
  *(v8h*)lds = *(const v8h*)g;
#endif
}

__device__ __forceinline__ void cp_fence() {
#ifdef HAS_ASYNC_LDS
  __builtin_amdgcn_s_wait_asynccnt(0);
#endif
}

// ---------------------------------------------------------------------------
// WMMA fragment loader (CDNA5 16x16x32 f16 layout, wave32).
// Tiles are staged in LDS with K contiguous (32 halves = 64B per row/col).
// Lane l (hi=l>>4) needs halves K=[hi*8..hi*8+7] and K=[16+hi*8..16+hi*8+7]:
// two aligned 16B chunks -> two ds_load_b128.
// C/D (16x16 f32): lane l -> col N = l&15; VGPR r -> row M = r + hi*8.
// ---------------------------------------------------------------------------
__device__ __forceinline__ v16h frag_load_k32(const _Float16* __restrict__ base,
                                              int idx, int hi) {
  const v8h* p = (const v8h*)(base + idx * 32);
  v8h lo = p[hi];
  v8h hh = p[2 + hi];
  return __builtin_shufflevector(lo, hh, 0, 1, 2, 3, 4, 5, 6, 7,
                                 8, 9, 10, 11, 12, 13, 14, 15);
}

// Stage one A tile [BM x BK] and one B tile [BN cols x BK k] from padded f16
// sources (A16 row-major ld=K, Wt col-major ld=K). Pure 16B copies.
__device__ __forceinline__ void stage_tiles(_Float16* __restrict__ Als,
                                            _Float16* __restrict__ Bls,
                                            const _Float16* __restrict__ A16,
                                            const _Float16* __restrict__ Wt,
                                            int row0, int col0, int kk, int K,
                                            int tid) {
  {  // A: 64 rows x 32 halves = 256 x 16B chunks, one per thread
    int r = tid >> 2, c8 = (tid & 3) * 8;
    cp16_g2l(Als + r * 32 + c8, A16 + (size_t)(row0 + r) * K + kk + c8);
  }
#pragma unroll
  for (int j = 0; j < 2; ++j) {  // B: 128 cols x 32 halves = 512 chunks
    int i = tid + j * 256;
    int c = i >> 2, c8 = (i & 3) * 8;
    cp16_g2l(Bls + c * 32 + c8, Wt + (size_t)(col0 + c) * K + kk + c8);
  }
}

// ---------------------------------------------------------------------------
// Generic WMMA GEMM: out[M,Nout] = act(A16[Mpad,K] @ Wt[Nout,K]^T + bias)
// Double-buffered LDS; grid = (Mpad/64, Nout/128), block = 256 (8 waves 4x2).
// ---------------------------------------------------------------------------
__global__ __launch_bounds__(256) void wmma_gemm_kernel(
    const _Float16* __restrict__ A16, const _Float16* __restrict__ Wt,
    const float* __restrict__ bias, float* __restrict__ out,
    int M, int K, int Nout, int relu) {
  __shared__ __attribute__((aligned(16))) _Float16 Als[2][BM * BK];
  __shared__ __attribute__((aligned(16))) _Float16 Bls[2][BN * BK];
  int tid = threadIdx.x, lane = tid & 31, wave = tid >> 5;
  int wrow = wave >> 1, wcol = wave & 1;
  int row0 = blockIdx.x * BM, col0 = blockIdx.y * BN;
  int n0 = lane & 15, hi = lane >> 4;
  int nk = K / BK;
  v8f acc[4] = {};
  stage_tiles(Als[0], Bls[0], A16, Wt, row0, col0, 0, K, tid);
  cp_fence();
  __syncthreads();
  for (int s = 0; s < nk; ++s) {
    int cur = s & 1;
    if (s + 1 < nk)
      stage_tiles(Als[cur ^ 1], Bls[cur ^ 1], A16, Wt, row0, col0,
                  (s + 1) * BK, K, tid);
    v16h af = frag_load_k32(Als[cur], wrow * 16 + n0, hi);
#pragma unroll
    for (int t = 0; t < 4; ++t) {
      v16h bf = frag_load_k32(Bls[cur], wcol * 64 + t * 16 + n0, hi);
      acc[t] = __builtin_amdgcn_wmma_f32_16x16x32_f16(
          false, af, false, bf, (short)0, acc[t], false, false);
    }
    cp_fence();
    __syncthreads();
  }
#pragma unroll
  for (int t = 0; t < 4; ++t) {
    int col = col0 + wcol * 64 + t * 16 + n0;
    float bv = bias ? bias[col] : 0.f;
#pragma unroll
    for (int r = 0; r < 8; ++r) {
      int row = row0 + wrow * 16 + hi * 8 + r;
      if (row < M) {
        float v = acc[t][r] + bv;
        if (relu) v = fmaxf(v, 0.f);
        out[(size_t)row * Nout + col] = v;
      }
    }
  }
}

// ---------------------------------------------------------------------------
// Fused gconv: out = relu(hin@ws + b + sum_k coeff[:,k] * (m @ wk[k]))
// 17 WMMA passes per tile; pass 0 uses hin16/wst (scale 1), passes 1..16 use
// m16/wkt[k] scaled per-row by coeff[row,k]. Avoids the 800MB [N,K,H] temp.
// ---------------------------------------------------------------------------
__global__ __launch_bounds__(256) void gconv_kernel(
    const _Float16* __restrict__ hin16, const _Float16* __restrict__ m16,
    const float* __restrict__ coeff, const _Float16* __restrict__ wst,
    const _Float16* __restrict__ wkt, const float* __restrict__ bias,
    float* __restrict__ out, int M) {
  __shared__ __attribute__((aligned(16))) _Float16 Als[2][BM * BK];
  __shared__ __attribute__((aligned(16))) _Float16 Bls[2][BN * BK];
  int tid = threadIdx.x, lane = tid & 31, wave = tid >> 5;
  int wrow = wave >> 1, wcol = wave & 1;
  int row0 = blockIdx.x * BM, col0 = blockIdx.y * BN;
  int n0 = lane & 15, hi = lane >> 4;
  v8f fin[4] = {};
  for (int pass = 0; pass < KKR + 1; ++pass) {
    const _Float16* Ap = (pass == 0) ? hin16 : m16;
    const _Float16* Wp = (pass == 0) ? wst : (wkt + (size_t)(pass - 1) * HH * HH);
    v8f acc[4] = {};
    stage_tiles(Als[0], Bls[0], Ap, Wp, row0, col0, 0, HH, tid);
    cp_fence();
    __syncthreads();
    for (int s = 0; s < HH / BK; ++s) {
      int cur = s & 1;
      if (s + 1 < HH / BK)
        stage_tiles(Als[cur ^ 1], Bls[cur ^ 1], Ap, Wp, row0, col0,
                    (s + 1) * BK, HH, tid);
      v16h af = frag_load_k32(Als[cur], wrow * 16 + n0, hi);
#pragma unroll
      for (int t = 0; t < 4; ++t) {
        v16h bf = frag_load_k32(Bls[cur], wcol * 64 + t * 16 + n0, hi);
        acc[t] = __builtin_amdgcn_wmma_f32_16x16x32_f16(
            false, af, false, bf, (short)0, acc[t], false, false);
      }
      cp_fence();
      __syncthreads();
    }
    if (pass == 0) {
#pragma unroll
      for (int t = 0; t < 4; ++t) fin[t] += acc[t];
    } else {
#pragma unroll
      for (int r = 0; r < 8; ++r) {
        int row = row0 + wrow * 16 + hi * 8 + r;
        float sc = (row < M) ? coeff[(size_t)row * AA + (pass - 1)] : 0.f;
#pragma unroll
        for (int t = 0; t < 4; ++t) fin[t][r] += sc * acc[t][r];
      }
    }
  }
#pragma unroll
  for (int t = 0; t < 4; ++t) {
    int col = col0 + wcol * 64 + t * 16 + n0;
    float bv = bias[col];
#pragma unroll
    for (int r = 0; r < 8; ++r) {
      int row = row0 + wrow * 16 + hi * 8 + r;
      if (row < M) out[(size_t)row * HH + col] = fmaxf(fin[t][r] + bv, 0.f);
    }
  }
}

// ---------------------------------------------------------------------------
// f32 -> padded f16 conversions
// ---------------------------------------------------------------------------
// A[M,K] f32 -> A16[Mpad,K] f16 (zero pad rows)
__global__ void convert_a_kernel(const float* __restrict__ A,
                                 _Float16* __restrict__ A16, int M, int K,
                                 long total) {
  long i = (long)blockIdx.x * 256 + threadIdx.x;
  if (i >= total) return;
  long row = i / K;
  A16[i] = (row < M) ? (_Float16)A[i] : (_Float16)0.f;
}

// W[nmat][K,Nout] f32 row-major -> Wt[nmat][Nout,K] f16 (K contiguous)
__global__ void convert_wt_kernel(const float* __restrict__ W,
                                  _Float16* __restrict__ Wt, int K, int Nout,
                                  long total) {
  long i = (long)blockIdx.x * 256 + threadIdx.x;
  if (i >= total) return;
  long mat = i / ((long)K * Nout);
  long rem = i - mat * (long)K * Nout;
  int r = (int)(rem / Nout), c = (int)(rem - (long)r * Nout);
  Wt[mat * (long)K * Nout + (long)c * K + r] = (_Float16)W[i];
}

// ---------------------------------------------------------------------------
// Utility / irregular kernels
// ---------------------------------------------------------------------------
__global__ void zero_kernel(float* p, long n) {
  long i = (long)blockIdx.x * 256 + threadIdx.x;
  if (i < n) p[i] = 0.f;
}

__global__ void scatter_feat_kernel(const float* __restrict__ x,
                                    const int* __restrict__ src,
                                    const int* __restrict__ dst,
                                    const float* __restrict__ emask,
                                    float* __restrict__ msg, int F) {
  int per = F >> 2;
  int gid = blockIdx.x * 256 + threadIdx.x;
  if (gid >= EE * per) return;
  int e = gid / per;
  int c = (gid - e * per) * 4;
  float w = emask ? emask[e] : 1.f;
  if (w == 0.f) return;
  const float* xp = x + (size_t)src[e] * F + c;
  float* mp = msg + (size_t)dst[e] * F + c;
  atomicAdd(mp + 0, xp[0] * w);
  atomicAdd(mp + 1, xp[1] * w);
  atomicAdd(mp + 2, xp[2] * w);
  atomicAdd(mp + 3, xp[3] * w);
}

__global__ void scatter_deg_kernel(const int* __restrict__ dst,
                                   const float* __restrict__ emask,
                                   float* __restrict__ deg) {
  int e = blockIdx.x * 256 + threadIdx.x;
  if (e >= EE) return;
  atomicAdd(&deg[dst[e]], emask ? emask[e] : 1.f);
}

__global__ void mean_div_kernel(float* __restrict__ msg,
                                const float* __restrict__ deg, int F, long n) {
  long i = (long)blockIdx.x * 256 + threadIdx.x;
  if (i >= n) return;
  msg[i] = msg[i] / fmaxf(deg[i / F], 1.f);
}

__global__ __launch_bounds__(256) void coeff_softmax_kernel(
    const float* __restrict__ h, const float* __restrict__ dict,
    float* __restrict__ coeff, int* __restrict__ assign) {
  __shared__ float dls[AA * HH];
  for (int i = threadIdx.x; i < AA * HH; i += 256) dls[i] = dict[i];
  __syncthreads();
  int n = blockIdx.x * 256 + threadIdx.x;
  if (n >= NN) return;
  const float* hr = h + (size_t)n * HH;
  float dots[AA];
#pragma unroll
  for (int a = 0; a < AA; ++a) dots[a] = 0.f;
  for (int j = 0; j < HH; ++j) {
    float hv = hr[j];
#pragma unroll
    for (int a = 0; a < AA; ++a) dots[a] += hv * dls[a * HH + j];
  }
  float mx = dots[0];
  int am = 0;
#pragma unroll
  for (int a = 1; a < AA; ++a)
    if (dots[a] > mx) { mx = dots[a]; am = a; }
  float se = 0.f;
#pragma unroll
  for (int a = 0; a < AA; ++a) { dots[a] = expf(dots[a] - mx); se += dots[a]; }
  float inv = 1.f / se;
#pragma unroll
  for (int a = 0; a < AA; ++a) coeff[(size_t)n * AA + a] = dots[a] * inv;
  if (assign) assign[n] = am;
}

__global__ void emask_kernel(const int* __restrict__ assign,
                             const int* __restrict__ src,
                             const int* __restrict__ dst,
                             float* __restrict__ emask) {
  int e = blockIdx.x * 256 + threadIdx.x;
  if (e >= EE) return;
  emask[e] = (assign[src[e]] == assign[dst[e]]) ? 1.f : 0.f;
}

__global__ void prep_pn_kernel(const float* __restrict__ p, float* __restrict__ pn) {
  __shared__ float red[256];
  int t = threadIdx.x;
  float v = p[t];
  red[t] = v * v;
  __syncthreads();
  for (int s = 128; s > 0; s >>= 1) {
    if (t < s) red[t] += red[t + s];
    __syncthreads();
  }
  pn[t] = v * rsqrtf(red[0]);
}

__global__ void pool_w_kernel(const float* __restrict__ x1, const float* __restrict__ x2,
                              const float* __restrict__ x3, const float* __restrict__ pn,
                              float* __restrict__ wout) {
  int gw = (blockIdx.x * 256 + threadIdx.x) >> 5;
  int lane = threadIdx.x & 31;
  if (gw >= NN) return;
  const float* r1 = x1 + (size_t)gw * HH;
  const float* r2 = x2 + (size_t)gw * HH;
  const float* r3 = x3 + (size_t)gw * HH;
  float d1 = 0, d2 = 0, d3 = 0;
  for (int j = lane; j < HH; j += 32) {
    float p = pn[j];
    d1 += r1[j] * p; d2 += r2[j] * p; d3 += r3[j] * p;
  }
  for (int off = 16; off; off >>= 1) {
    d1 += __shfl_xor(d1, off, 32);
    d2 += __shfl_xor(d2, off, 32);
    d3 += __shfl_xor(d3, off, 32);
  }
  if (lane == 0) {
    float mx = fmaxf(d1, fmaxf(d2, d3));
    float e1 = expf(d1 - mx), e2 = expf(d2 - mx), e3 = expf(d3 - mx);
    float inv = 1.f / (e1 + e2 + e3);
    wout[gw * 3 + 0] = e1 * inv;
    wout[gw * 3 + 1] = e2 * inv;
    wout[gw * 3 + 2] = e3 * inv;
  }
}

__global__ void blend_kernel(const float* __restrict__ x1, const float* __restrict__ x2,
                             const float* __restrict__ x3, const float* __restrict__ w,
                             float* __restrict__ xm) {
  long i = (long)blockIdx.x * 256 + threadIdx.x;
  if (i >= (long)NN * HH) return;
  long n = i >> 8;
  xm[i] = w[n * 3 + 0] * x1[i] + w[n * 3 + 1] * x2[i] + w[n * 3 + 2] * x3[i];
}

__device__ __forceinline__ void atomicMaxFloat(float* addr, float val) {
  unsigned int* ua = (unsigned int*)addr;
  unsigned int old = *ua;
  while (true) {
    float f = __uint_as_float(old);
    if (f >= val) break;
    unsigned int assumed = old;
    old = atomicCAS(ua, assumed, __float_as_uint(val));
    if (old == assumed) break;
  }
}

__global__ void init_att_kernel(float* __restrict__ smax, float* __restrict__ den) {
  int i = blockIdx.x * 256 + threadIdx.x;
  if (i >= BB * NHD) return;
  smax[i] = -1e30f;
  den[i] = 0.f;
}

__global__ void att_s_kernel(const float* __restrict__ xm, const float* __restrict__ wa,
                             const int* __restrict__ batch, float* __restrict__ s,
                             float* __restrict__ smax) {
  int gw = (blockIdx.x * 256 + threadIdx.x) >> 5;
  int lane = threadIdx.x & 31;
  if (gw >= NN) return;
  const float* xr = xm + (size_t)gw * HH;
  float a0 = 0, a1 = 0, a2 = 0, a3 = 0;
  for (int j = lane; j < HH; j += 32) {
    float v = xr[j];
    a0 += v * wa[j * NHD + 0];
    a1 += v * wa[j * NHD + 1];
    a2 += v * wa[j * NHD + 2];
    a3 += v * wa[j * NHD + 3];
  }
  for (int off = 16; off; off >>= 1) {
    a0 += __shfl_xor(a0, off, 32);
    a1 += __shfl_xor(a1, off, 32);
    a2 += __shfl_xor(a2, off, 32);
    a3 += __shfl_xor(a3, off, 32);
  }
  if (lane == 0) {
    int b = batch[gw];
    s[gw * NHD + 0] = a0; atomicMaxFloat(&smax[b * NHD + 0], a0);
    s[gw * NHD + 1] = a1; atomicMaxFloat(&smax[b * NHD + 1], a1);
    s[gw * NHD + 2] = a2; atomicMaxFloat(&smax[b * NHD + 2], a2);
    s[gw * NHD + 3] = a3; atomicMaxFloat(&smax[b * NHD + 3], a3);
  }
}

__global__ void att_e_kernel(const float* __restrict__ s, const float* __restrict__ smax,
                             const int* __restrict__ batch, float* __restrict__ e,
                             float* __restrict__ den) {
  int i = blockIdx.x * 256 + threadIdx.x;
  if (i >= NN * NHD) return;
  int n = i >> 2, h = i & 3;
  int b = batch[n];
  float ev = expf(s[i] - smax[b * NHD + h]);
  e[i] = ev;
  atomicAdd(&den[b * NHD + h], ev);
}

__global__ void att_o_kernel(const float* __restrict__ e, const float* __restrict__ den,
                             const int* __restrict__ batch, const float* __restrict__ v,
                             float* __restrict__ o) {
  long i = (long)blockIdx.x * 256 + threadIdx.x;
  if (i >= (long)NN * HH) return;
  long n = i >> 8;
  int col = (int)(i & 255);
  int head = col >> 6;  // H/NH = 64
  int b = batch[n];
  float a = e[n * NHD + head] / den[b * NHD + head];
  atomicAdd(&o[(size_t)b * HH + col], a * v[i]);
}

__global__ __launch_bounds__(256) void cls_kernel(const float* __restrict__ t,
                                                  const float* __restrict__ w2,
                                                  const float* __restrict__ b2,
                                                  float* __restrict__ logits) {
  __shared__ float w2s[HH * CC];
  for (int i = threadIdx.x; i < HH * CC; i += 256) w2s[i] = w2[i];
  __syncthreads();
  int b = blockIdx.x * 256 + threadIdx.x;
  if (b >= BB) return;
  const float* tr = t + (size_t)b * HH;
  float z[CC];
#pragma unroll
  for (int c = 0; c < CC; ++c) z[c] = b2[c];
  for (int j = 0; j < HH; ++j) {
    float tv = tr[j];
#pragma unroll
    for (int c = 0; c < CC; ++c) z[c] += tv * w2s[j * CC + c];
  }
  float mx = z[0];
#pragma unroll
  for (int c = 1; c < CC; ++c) mx = fmaxf(mx, z[c]);
  float se = 0.f;
#pragma unroll
  for (int c = 0; c < CC; ++c) se += expf(z[c] - mx);
  float ls = mx + logf(se);
#pragma unroll
  for (int c = 0; c < CC; ++c) logits[(size_t)b * CC + c] = z[c] - ls;
}

// ---------------------------------------------------------------------------
// Host launcher
// ---------------------------------------------------------------------------
static inline int nblk(long n) { return (int)((n + 255) / 256); }

extern "C" void kernel_launch(void* const* d_in, const int* in_sizes, int n_in,
                              void* d_out, int out_size, void* d_ws, size_t ws_size,
                              hipStream_t stream) {
  const float* x       = (const float*)d_in[0];
  const int*   ei      = (const int*)d_in[1];
  const int*   src     = ei;
  const int*   dst     = ei + EE;
  const int*   batch   = (const int*)d_in[2];
  const float* enc_w0  = (const float*)d_in[3];
  const float* enc_b0  = (const float*)d_in[4];
  const float* enc_w1  = (const float*)d_in[5];
  const float* enc_b1  = (const float*)d_in[6];
  const float* dict    = (const float*)d_in[7];
  const float* wk1     = (const float*)d_in[8];
  const float* ws1     = (const float*)d_in[9];
  const float* b1      = (const float*)d_in[10];
  const float* wk2     = (const float*)d_in[11];
  const float* ws2     = (const float*)d_in[12];
  const float* b2      = (const float*)d_in[13];
  const float* wk3     = (const float*)d_in[14];
  const float* ws3     = (const float*)d_in[15];
  const float* b3      = (const float*)d_in[16];
  const float* score_p = (const float*)d_in[17];
  const float* att_wa  = (const float*)d_in[18];
  const float* att_wv  = (const float*)d_in[19];
  const float* att_wo  = (const float*)d_in[20];
  const float* cls_w1  = (const float*)d_in[21];
  const float* cls_b1  = (const float*)d_in[22];
  const float* cls_w2  = (const float*)d_in[23];
  const float* cls_b2  = (const float*)d_in[24];

  // Workspace layout (float units; every block stays 16B aligned)
  float* W = (float*)d_ws;
  size_t off = 0;
  float* deg   = W + off; off += NN;
  float* degm  = W + off; off += NN + 16;  // keep 16B alignment downstream
  float* msg   = W + off; off += (size_t)NN * HH;  // agg output, reused as v
  float* h     = W + off; off += (size_t)NN * HH;
  float* x1    = W + off; off += (size_t)NN * HH;
  float* x2    = W + off; off += (size_t)NN * HH;
  float* x3    = W + off; off += (size_t)NN * HH;
  float* xm    = W + off; off += (size_t)NN * HH;
  float* coeff = W + off; off += (size_t)NN * AA;
  int*   assign = (int*)(W + off); off += NN;
  float* emask = W + off; off += EE;
  float* pn    = W + off; off += HH;
  float* wbl   = W + off; off += (size_t)NN * 3;
  float* sb    = W + off; off += (size_t)NN * NHD;
  float* eb    = W + off; off += (size_t)NN * NHD;
  float* smax  = W + off; off += BB * NHD;
  float* den   = W + off; off += BB * NHD + 8;
  float* obuf  = W + off; off += (size_t)BB * HH;
  float* tbuf  = W + off; off += (size_t)BB * HH;
  // f16 regions (2 halves per float slot)
  _Float16* a16a = (_Float16*)(W + off); off += (size_t)NPAD * HH / 2;
  _Float16* a16b = (_Float16*)(W + off); off += (size_t)NPAD * HH / 2;
  _Float16* w16  = (_Float16*)(W + off);
  size_t hoff = 0;
  _Float16* enc_w0t = w16 + hoff; hoff += (size_t)FF * HH;
  _Float16* enc_w1t = w16 + hoff; hoff += (size_t)HH * HH;
  _Float16* ws1t    = w16 + hoff; hoff += (size_t)HH * HH;
  _Float16* ws2t    = w16 + hoff; hoff += (size_t)HH * HH;
  _Float16* ws3t    = w16 + hoff; hoff += (size_t)HH * HH;
  _Float16* wk1t    = w16 + hoff; hoff += (size_t)KKR * HH * HH;
  _Float16* wk2t    = w16 + hoff; hoff += (size_t)KKR * HH * HH;
  _Float16* wk3t    = w16 + hoff; hoff += (size_t)KKR * HH * HH;
  _Float16* att_wvt = w16 + hoff; hoff += (size_t)HH * HH;
  _Float16* att_wot = w16 + hoff; hoff += (size_t)HH * HH;
  _Float16* cls_w1t = w16 + hoff; hoff += (size_t)HH * HH;

  float* logits = (float*)d_out;
  float* g      = logits + (size_t)BB * CC;  // second output, contiguous

  dim3 ggemmN(NPAD / BM, HH / BN);
  dim3 ggemmB(BPAD / BM, HH / BN);
  const long NHtot = (long)NPAD * HH;
  const long BHtot = (long)BPAD * HH;

  // ---- One-time weight conversion (f32 row-major -> f16 [N][K])
  convert_wt_kernel<<<nblk((long)FF * HH), 256, 0, stream>>>(enc_w0, enc_w0t, FF, HH, (long)FF * HH);
  convert_wt_kernel<<<nblk((long)HH * HH), 256, 0, stream>>>(enc_w1, enc_w1t, HH, HH, (long)HH * HH);
  convert_wt_kernel<<<nblk((long)HH * HH), 256, 0, stream>>>(ws1, ws1t, HH, HH, (long)HH * HH);
  convert_wt_kernel<<<nblk((long)HH * HH), 256, 0, stream>>>(ws2, ws2t, HH, HH, (long)HH * HH);
  convert_wt_kernel<<<nblk((long)HH * HH), 256, 0, stream>>>(ws3, ws3t, HH, HH, (long)HH * HH);
  convert_wt_kernel<<<nblk((long)KKR * HH * HH), 256, 0, stream>>>(wk1, wk1t, HH, HH, (long)KKR * HH * HH);
  convert_wt_kernel<<<nblk((long)KKR * HH * HH), 256, 0, stream>>>(wk2, wk2t, HH, HH, (long)KKR * HH * HH);
  convert_wt_kernel<<<nblk((long)KKR * HH * HH), 256, 0, stream>>>(wk3, wk3t, HH, HH, (long)KKR * HH * HH);
  convert_wt_kernel<<<nblk((long)HH * HH), 256, 0, stream>>>(att_wv, att_wvt, HH, HH, (long)HH * HH);
  convert_wt_kernel<<<nblk((long)HH * HH), 256, 0, stream>>>(att_wo, att_wot, HH, HH, (long)HH * HH);
  convert_wt_kernel<<<nblk((long)HH * HH), 256, 0, stream>>>(cls_w1, cls_w1t, HH, HH, (long)HH * HH);

  // ---- Encoder layer 1: mean agg (F=128) + linear + relu
  zero_kernel<<<nblk((long)NN * FF), 256, 0, stream>>>(msg, (long)NN * FF);
  zero_kernel<<<nblk(NN), 256, 0, stream>>>(deg, NN);
  scatter_deg_kernel<<<nblk(EE), 256, 0, stream>>>(dst, nullptr, deg);
  scatter_feat_kernel<<<nblk((long)EE * (FF / 4)), 256, 0, stream>>>(x, src, dst, nullptr, msg, FF);
  mean_div_kernel<<<nblk((long)NN * FF), 256, 0, stream>>>(msg, deg, FF, (long)NN * FF);
  convert_a_kernel<<<nblk((long)NPAD * FF), 256, 0, stream>>>(msg, a16a, NN, FF, (long)NPAD * FF);
  wmma_gemm_kernel<<<ggemmN, 256, 0, stream>>>(a16a, enc_w0t, enc_b0, h, NN, FF, HH, 1);

  // ---- Encoder layer 2 (deg unchanged: all-ones mask)
  zero_kernel<<<nblk((long)NN * HH), 256, 0, stream>>>(msg, (long)NN * HH);
  scatter_feat_kernel<<<nblk((long)EE * (HH / 4)), 256, 0, stream>>>(h, src, dst, nullptr, msg, HH);
  mean_div_kernel<<<nblk((long)NN * HH), 256, 0, stream>>>(msg, deg, HH, (long)NN * HH);
  convert_a_kernel<<<nblk(NHtot), 256, 0, stream>>>(msg, a16a, NN, HH, NHtot);
  wmma_gemm_kernel<<<ggemmN, 256, 0, stream>>>(a16a, enc_w1t, enc_b1, h, NN, HH, HH, 1);

  // ---- Dictionary assignment -> edge mask, masked degrees
  coeff_softmax_kernel<<<nblk(NN), 256, 0, stream>>>(h, dict, coeff, assign);
  emask_kernel<<<nblk(EE), 256, 0, stream>>>(assign, src, dst, emask);
  zero_kernel<<<nblk(NN), 256, 0, stream>>>(degm, NN);
  scatter_deg_kernel<<<nblk(EE), 256, 0, stream>>>(dst, emask, degm);

  // ---- 3 fused gconv layers
  const _Float16* wkts[3] = {wk1t, wk2t, wk3t};
  const _Float16* wsts[3] = {ws1t, ws2t, ws3t};
  const float* bs[3]  = {b1, b2, b3};
  const float* hins[3] = {h, x1, x2};
  float* xouts[3] = {x1, x2, x3};
  for (int L = 0; L < 3; ++L) {
    coeff_softmax_kernel<<<nblk(NN), 256, 0, stream>>>(hins[L], dict, coeff, nullptr);
    zero_kernel<<<nblk((long)NN * HH), 256, 0, stream>>>(msg, (long)NN * HH);
    scatter_feat_kernel<<<nblk((long)EE * (HH / 4)), 256, 0, stream>>>(hins[L], src, dst, emask, msg, HH);
    mean_div_kernel<<<nblk((long)NN * HH), 256, 0, stream>>>(msg, degm, HH, (long)NN * HH);
    convert_a_kernel<<<nblk(NHtot), 256, 0, stream>>>(hins[L], a16a, NN, HH, NHtot);
    convert_a_kernel<<<nblk(NHtot), 256, 0, stream>>>(msg, a16b, NN, HH, NHtot);
    gconv_kernel<<<ggemmN, 256, 0, stream>>>(a16a, a16b, coeff, wsts[L], wkts[L], bs[L], xouts[L], NN);
  }

  // ---- Score-weighted blend
  prep_pn_kernel<<<1, 256, 0, stream>>>(score_p, pn);
  pool_w_kernel<<<nblk((long)NN * 32), 256, 0, stream>>>(x1, x2, x3, pn, wbl);
  blend_kernel<<<nblk((long)NN * HH), 256, 0, stream>>>(x1, x2, x3, wbl, xm);

  // ---- Multi-head attention graph pooling
  init_att_kernel<<<nblk(BB * NHD), 256, 0, stream>>>(smax, den);
  att_s_kernel<<<nblk((long)NN * 32), 256, 0, stream>>>(xm, att_wa, batch, sb, smax);
  att_e_kernel<<<nblk((long)NN * NHD), 256, 0, stream>>>(sb, smax, batch, eb, den);
  convert_a_kernel<<<nblk(NHtot), 256, 0, stream>>>(xm, a16a, NN, HH, NHtot);
  wmma_gemm_kernel<<<ggemmN, 256, 0, stream>>>(a16a, att_wvt, nullptr, msg /*v*/, NN, HH, HH, 0);
  zero_kernel<<<nblk((long)BB * HH), 256, 0, stream>>>(obuf, (long)BB * HH);
  att_o_kernel<<<nblk((long)NN * HH), 256, 0, stream>>>(eb, den, batch, msg, obuf);
  convert_a_kernel<<<nblk(BHtot), 256, 0, stream>>>(obuf, a16a, BB, HH, BHtot);
  wmma_gemm_kernel<<<ggemmB, 256, 0, stream>>>(a16a, att_wot, nullptr, g, BB, HH, HH, 0);

  // ---- Classifier
  convert_a_kernel<<<nblk(BHtot), 256, 0, stream>>>(g, a16a, BB, HH, BHtot);
  wmma_gemm_kernel<<<ggemmB, 256, 0, stream>>>(a16a, cls_w1t, cls_b1, tbuf, BB, HH, HH, 1);
  cls_kernel<<<nblk(BB), 256, 0, stream>>>(tbuf, cls_w2, cls_b2, logits);
}
